// _PaiNNMessageBlock_79663053406272
// MI455X (gfx1250) — matrix operand order repaired and checked by reading, not compiled, not run on hardware
//
#include <hip/hip_runtime.h>
#include <hip/hip_bf16.h>

// ---------------- problem constants ----------------
#define NN      50000
#define FF      128
#define RBF_K   20
#define EE      800000
#define NTILES  (EE / 16)          // 50000 exact
#define F3      (3 * FF)           // 384

static_assert(EE % 16 == 0, "edge count must be multiple of 16");

typedef __bf16 bf16;
typedef __attribute__((ext_vector_type(16))) __bf16 v16bf;
typedef __attribute__((ext_vector_type(8)))  __bf16 v8bf;
typedef __attribute__((ext_vector_type(8)))  float  v8f;

union FragU { v16bf v; v8bf h[2]; };

__device__ __forceinline__ v16bf load_frag(const bf16* p0, const bf16* p1) {
    FragU u;
    u.h[0] = *(const v8bf*)p0;   // 16-byte load (ds_load_b128)
    u.h[1] = *(const v8bf*)p1;
    return u.v;
}

__device__ __forceinline__ unsigned pack_bf16(float a, float b) {
    union { bf16 h[2]; unsigned u; } p;
    p.h[0] = (bf16)a; p.h[1] = (bf16)b;    // lowers to v_cvt_pk_bf16_f32
    return p.u;
}

__device__ __forceinline__ v8f wmma_bf16(v16bf a, v16bf b, v8f c) {
    return __builtin_amdgcn_wmma_f32_16x16x32_bf16(
        /*neg_a=*/false, a, /*neg_b=*/false, b,
        /*c_mod=*/(short)0, c, /*reuse_a=*/false, /*reuse_b=*/false);
}

// Raw hardware transcendentals. v_exp_f32/v_log_f32 are TRANS ops: one slot is
// required before the result may be consumed (ISA 7.4.3), so embed a v_nop.
__device__ __forceinline__ float hw_exp2(float x) {
    float r;
    asm volatile("v_exp_f32 %0, %1\n\tv_nop" : "=v"(r) : "v"(x));
    return r;
}
__device__ __forceinline__ float hw_log2(float x) {
    float r;
    asm volatile("v_log_f32 %0, %1\n\tv_nop" : "=v"(r) : "v"(x));
    return r;
}

// softplus(x) - ln2 == max(x,0) + ln2*log2(0.5 + 0.5*exp2(-|x|*log2e))
__device__ __forceinline__ float softplus_m_ln2(float x) {
    const float LOG2E = 1.4426950408889634f;
    const float LN2   = 0.6931471805599453f;
    float mx = fmaxf(x, 0.0f);
    float t  = hw_exp2(-fabsf(x) * LOG2E);     // in (0,1]
    float l  = hw_log2(fmaf(0.5f, t, 0.5f));   // arg in (0.5,1]
    return fmaf(LN2, l, mx);
}

// ---------------- ws layout (bf16 elements) ----------------
// f1Wp : [128 n][32 k]  (k >= 20 zero padded)
// f2Wt : [384 n][128 k]
// phiWt: [384 n][128 k]
#define WS_F1_OFF   0
#define WS_F2_OFF   4096
#define WS_PHI_OFF  (4096 + 49152)
#define WS_TOTAL_ELEMS (4096 + 49152 + 49152)   // 102400 bf16
#define WS_TOTAL_BYTES (WS_TOTAL_ELEMS * 2)     // 204800 B

// ---------------- LDS layout (bytes) ----------------
// [0 .. 204800)        staged weights (same layout as ws)
// [204800 .. 208384)   biases: f1_b[128] | f2_b[384] | phi_b[384] (f32)
// then per-wave region (9536 B each, 8 waves):
//   +0    : eT   bf16[16][32]   (1024 B)
//   +1024 : hT   bf16[16][128]  (4096 B)
//   +5120 : sT   bf16[16][128]  (4096 B)
//   +9216 : rT   float[16][3]   ( 192 B)
//   +9408 : srcT int[16]        (  64 B)
//   +9472 : dstT int[16]        (  64 B)
#define LDS_BIAS_OFF   WS_TOTAL_BYTES
#define LDS_WAVE_OFF   (WS_TOTAL_BYTES + 3584)
#define LDS_WAVE_BYTES 9536
#define NWAVES 8
#define LDS_TOTAL (LDS_WAVE_OFF + NWAVES * LDS_WAVE_BYTES)   // 284672 B

__global__ void painn_zero_kernel(float* __restrict__ p, int n) {
    for (int i = blockIdx.x * blockDim.x + threadIdx.x; i < n;
         i += gridDim.x * blockDim.x)
        p[i] = 0.0f;
}

// Transpose + convert weights into ws (bf16, output-column-major rows of K).
__global__ void painn_prep_kernel(const float* __restrict__ phiW,
                                  const float* __restrict__ f1W,
                                  const float* __restrict__ f2W,
                                  bf16* __restrict__ ws) {
    int i = blockIdx.x * blockDim.x + threadIdx.x;
    if (i < 4096) {                         // f1Wp[n][k], k padded to 32
        int n = i >> 5, k = i & 31;
        float v = (k < RBF_K) ? f1W[k * FF + n] : 0.0f;   // f1W is (20,128)
        ws[WS_F1_OFF + i] = (bf16)v;
    }
    int j = i - 4096;
    if (j >= 0 && j < 49152) {              // f2Wt[n][k] from f2W (128,384)
        int n = j >> 7, k = j & 127;
        ws[WS_F2_OFF + j] = (bf16)f2W[k * F3 + n];
    }
    int l = i - (4096 + 49152);
    if (l >= 0 && l < 49152) {              // phiWt[n][k] from phiW (128,384)
        int n = l >> 7, k = l & 127;
        ws[WS_PHI_OFF + l] = (bf16)phiW[k * F3 + n];
    }
}

__global__ __launch_bounds__(256)
void painn_edge_kernel(const float* __restrict__ s_in,
                       const float* __restrict__ v_in,
                       const float* __restrict__ e_ij,
                       const float* __restrict__ r_hat,
                       const float* __restrict__ phi_b,
                       const float* __restrict__ f1_b,
                       const float* __restrict__ f2_b,
                       const int*   __restrict__ eidx,   // [2][E]: row0=src, row1=dst
                       const bf16*  __restrict__ wsw,    // prepped weights
                       float* __restrict__ dS,           // (N,F)
                       float* __restrict__ dV) {         // (N,F,3)
    extern __shared__ char smem[];
    const int tid  = threadIdx.x;
    const int wave = tid >> 5;
    const int lane = tid & 31;

    // ---- stage weights (204800 B) global -> LDS via async copy ----
    {
        const unsigned ldsbase = (unsigned)(uintptr_t)smem;  // LDS byte offset
        for (int i = tid; i < WS_TOTAL_BYTES / 16; i += 256) {
            unsigned loff = ldsbase + (unsigned)i * 16u;
            unsigned goff = (unsigned)i * 16u;
            // ASYNCcnt-tracked DMA: LDS[loff] = MEM[wsw + goff], 16 B per lane
            asm volatile("global_load_async_to_lds_b128 %0, %1, %2"
                         :: "v"(loff), "v"(goff), "s"(wsw)
                         : "memory");
        }
    }
    // ---- stage biases (896 floats) ----
    {
        float* lb = (float*)(smem + LDS_BIAS_OFF);
        for (int i = tid; i < 896; i += 256) {
            float v;
            if (i < 128)      v = f1_b[i];
            else if (i < 512) v = f2_b[i - 128];
            else              v = phi_b[i - 512];
            lb[i] = v;
        }
    }
    asm volatile("s_wait_asynccnt 0x0" ::: "memory");
    __syncthreads();

    const bf16* lf1  = (const bf16*)smem + WS_F1_OFF;
    const bf16* lf2  = (const bf16*)smem + WS_F2_OFF;
    const bf16* lphi = (const bf16*)smem + WS_PHI_OFF;
    const float* lf1b = (const float*)(smem + LDS_BIAS_OFF);
    const float* lf2b = lf1b + 128;
    const float* lphb = lf1b + 512;

    char* wbase = smem + LDS_WAVE_OFF + wave * LDS_WAVE_BYTES;
    bf16*  eT   = (bf16*)(wbase);
    bf16*  hT   = (bf16*)(wbase + 1024);
    bf16*  sT   = (bf16*)(wbase + 5120);
    float* rT   = (float*)(wbase + 9216);
    int*   srcT = (int*)  (wbase + 9408);
    int*   dstT = (int*)  (wbase + 9472);

    const int row  = lane & 15;
    const int half = lane >> 4;
    const int stride = gridDim.x * NWAVES;

    for (int tile = blockIdx.x * NWAVES + wave; tile < NTILES; tile += stride) {
        const int e0 = tile * 16;

        // prefetch next tile's e_ij (covers 1280 B with 32 lanes x 40 B)
        {
            int tnext = tile + stride;
            if (tnext < NTILES)
                __builtin_prefetch(&e_ij[(size_t)tnext * 16 * RBF_K + lane * 10], 0, 1);
        }

        // ---- stage indices + r_hat for this 16-edge tile ----
        if (lane < 16) {
            srcT[lane] = eidx[e0 + lane];
            dstT[lane] = eidx[EE + e0 + lane];
            rT[lane * 3 + 0] = r_hat[(size_t)(e0 + lane) * 3 + 0];
            rT[lane * 3 + 1] = r_hat[(size_t)(e0 + lane) * 3 + 1];
            rT[lane * 3 + 2] = r_hat[(size_t)(e0 + lane) * 3 + 2];
        }

        // ---- stage e_ij tile as bf16 pairs, K padded 20 -> 32 ----
        {
            unsigned* eTp = (unsigned*)eT;   // 256 bf16-pairs
            #pragma unroll
            for (int idx = lane; idx < 256; idx += 32) {
                int m = idx >> 4, p = idx & 15;
                unsigned pk = 0u;
                if (p < RBF_K / 2) {
                    const float* ep = e_ij + (size_t)(e0 + m) * RBF_K + 2 * p;
                    pk = pack_bf16(ep[0], ep[1]);
                }
                eTp[idx] = pk;
            }
        }

        // ---- gather s[dst] rows -> bf16 pairs in LDS ----
        {
            unsigned* sTp = (unsigned*)sT;   // 1024 bf16-pairs
            #pragma unroll 4
            for (int idx = lane; idx < 1024; idx += 32) {
                int m = idx >> 6, p = idx & 63;
                const float2 fv = ((const float2*)(s_in + (size_t)dstT[m] * FF))[p];
                sTp[idx] = pack_bf16(fv.x, fv.y);
            }
        }

        // ---- h = softplus(e_ij @ f1_W + f1_b) - ln2  (16 x 128) ----
        {
            const bf16* ap0 = eT + row * 32 + half * 16;
            v16bf aE = load_frag(ap0, ap0 + 8);
            for (int t = 0; t < 8; ++t) {
                const int n = t * 16 + row;           // C-layout: N = lane%16
                const bf16* bp = lf1 + n * 32 + half * 16;
                v16bf b = load_frag(bp, bp + 8);
                v8f acc = {};
                acc = wmma_bf16(aE, b, acc);
                float bias = lf1b[n];
                #pragma unroll
                for (int j = 0; j < 8; ++j) {
                    float h = softplus_m_ln2(acc[j] + bias);
                    hT[(j + half * 8) * FF + n] = (bf16)h;
                }
            }
        }

        // ---- hoist per-edge scatter state into registers ----
        int   dSb[8], dVb[8], vb[8];
        float rr0[8], rr1[8], rr2[8];
        #pragma unroll
        for (int j = 0; j < 8; ++j) {
            int m  = j + half * 8;
            int si = srcT[m];
            int di = dstT[m];
            dSb[j] = si * FF;
            dVb[j] = si * F3;
            vb[j]  = di * F3;
            rr0[j] = rT[m * 3 + 0];
            rr1[j] = rT[m * 3 + 1];
            rr2[j] = rT[m * 3 + 2];
        }

        // ---- main fused GEMMs + epilogue, per 16-feature tile ----
        for (int t = 0; t < 8; ++t) {
            v8f aFs = {}, aFv = {}, aFr = {}, aPs = {}, aPr = {};
            #pragma unroll
            for (int kk = 0; kk < 4; ++kk) {
                const bf16* hp = hT + row * FF + kk * 32 + half * 16;
                v16bf aH = load_frag(hp, hp + 8);
                const bf16* sp = sT + row * FF + kk * 32 + half * 16;
                v16bf aS = load_frag(sp, sp + 8);
                const int kb = kk * 32 + half * 16;
                const bf16* b1 = lf2  + (size_t)((t     ) * 16 + row) * FF + kb; // W_s
                const bf16* b2 = lf2  + (size_t)((t +  8) * 16 + row) * FF + kb; // W_v
                const bf16* b3 = lf2  + (size_t)((t + 16) * 16 + row) * FF + kb; // W_r
                const bf16* b4 = lphi + (size_t)((t     ) * 16 + row) * FF + kb; // phi_s
                const bf16* b5 = lphi + (size_t)((t + 16) * 16 + row) * FF + kb; // phi_r
                aFs = wmma_bf16(aH, load_frag(b1, b1 + 8), aFs);
                aFv = wmma_bf16(aH, load_frag(b2, b2 + 8), aFv);
                aFr = wmma_bf16(aH, load_frag(b3, b3 + 8), aFr);
                aPs = wmma_bf16(aS, load_frag(b4, b4 + 8), aPs);
                aPr = wmma_bf16(aS, load_frag(b5, b5 + 8), aPr);
            }
            const int f   = t * 16 + row;              // global feature index
            const int f3o = f * 3;
            const float bFs = lf2b[f];
            const float bFv = lf2b[FF + f];
            const float bFr = lf2b[2 * FF + f];
            const float bPs = lphb[f];
            const float bPr = lphb[2 * FF + f];
            #pragma unroll
            for (int j = 0; j < 8; ++j) {
                float Ws = aFs[j] + bFs;
                float Wv = aFv[j] + bFv;
                float Wr = aFr[j] + bFr;
                float Ps = aPs[j] + bPs;
                float Pr = aPr[j] + bPr;
                atomicAdd(dS + dSb[j] + f, Ws * Ps);
                float wrpr = Wr * Pr;
                const float* vp  = v_in + vb[j] + f3o;
                float*       dvp = dV + dVb[j] + f3o;
                atomicAdd(dvp + 0, Wv * vp[0] + wrpr * rr0[j]);
                atomicAdd(dvp + 1, Wv * vp[1] + wrpr * rr1[j]);
                atomicAdd(dvp + 2, Wv * vp[2] + wrpr * rr2[j]);
            }
        }
    }
}

extern "C" void kernel_launch(void* const* d_in, const int* in_sizes, int n_in,
                              void* d_out, int out_size, void* d_ws, size_t ws_size,
                              hipStream_t stream) {
    const float* s_in   = (const float*)d_in[0];
    const float* v_in   = (const float*)d_in[1];
    const float* e_ij   = (const float*)d_in[2];
    const float* r_hat  = (const float*)d_in[3];
    const float* phi_W  = (const float*)d_in[4];
    const float* phi_b  = (const float*)d_in[5];
    const float* f1_W   = (const float*)d_in[6];
    const float* f1_b   = (const float*)d_in[7];
    const float* f2_W   = (const float*)d_in[8];
    const float* f2_b   = (const float*)d_in[9];
    const int*   eidx   = (const int*)d_in[10];

    float* out = (float*)d_out;
    float* dS  = out;
    float* dV  = out + (size_t)NN * FF;
    bf16*  ws  = (bf16*)d_ws;

    // 1) zero segment-sum outputs
    painn_zero_kernel<<<2048, 256, 0, stream>>>(out, out_size);

    // 2) weight transpose + bf16 convert into workspace
    painn_prep_kernel<<<(WS_TOTAL_ELEMS + 255) / 256, 256, 0, stream>>>(
        phi_W, f1_W, f2_W, ws);

    // 3) fused edge kernel: filter-net + phi gather-GEMM + message + scatter
    painn_edge_kernel<<<1024, 256, LDS_TOTAL, stream>>>(
        s_in, v_in, e_ij, r_hat, phi_b, f1_b, f2_b, eidx, ws, dS, dV);
}
